// WindowAttention_86260123172925
// MI455X (gfx1250) — matrix operand order repaired and checked by reading, not compiled, not run on hardware
//
#include <hip/hip_runtime.h>

typedef __attribute__((ext_vector_type(16))) __bf16 v16bf;
typedef __attribute__((ext_vector_type(8)))  __bf16 v8bf;
typedef __attribute__((ext_vector_type(8)))  float  v8f;

#define WMMA_BF16(A, B, C) \
  __builtin_amdgcn_wmma_f32_16x16x32_bf16(false, (A), false, (B), (short)0, (C), false, false)

// problem dims
constexpr int NTOK  = 49;
constexpr int CDIM  = 256;
constexpr int HEADS = 8;
constexpr int HD    = 32;
constexpr int NW    = 64;
constexpr int BWIN  = 2048;
constexpr float SCALE = 0.17677669529663687f;   // 32^-0.5

// LDS layout (bf16 element offsets); strides padded so ds_load_b128 stays 16B aligned
constexpr int XSTR   = 272;                      // 544 B rows (16B multiple)
constexpr int X_ELEMS = 64 * XSTR;               // 17408
constexpr int VTSTR  = 72;                       // 144 B rows (16B multiple)
constexpr int HQ  = 64 * HD;                     // q  [64][32]
constexpr int HK  = 64 * HD;                     // k  [64][32]
constexpr int HVT = HD * VTSTR;                  // vT [32][72]
constexpr int HEAD_ELEMS = HQ + HK + HVT;        // 6400
constexpr int OBSTR  = 264;                      // 528 B rows (16B multiple)
constexpr int OFF_X = 0;
constexpr int OFF_H = OFF_X + X_ELEMS;
constexpr int OFF_P = OFF_H + HEADS * HEAD_ELEMS;
constexpr int OFF_O = OFF_P + HEADS * 16 * 64;
constexpr int LDS_ELEMS = OFF_O + 64 * OBSTR;    // 93696 elems
constexpr size_t LDS_BYTES = (size_t)LDS_ELEMS * 2;  // 187392 B < 320 KB

__device__ __forceinline__ v16bf frag_split(const __bf16* lo, const __bf16* hi) {
  v16bf r;
  *(v8bf*)&r       = *(const v8bf*)lo;
  *((v8bf*)&r + 1) = *(const v8bf*)hi;
  return r;
}

// -------- prep: bf16-transposed weights into workspace --------
__global__ __launch_bounds__(256) void prep_kernel(const float* __restrict__ qkv_w,
                                                   const float* __restrict__ proj_w,
                                                   __bf16* __restrict__ wqT,
                                                   __bf16* __restrict__ wpT) {
  const int total = 768 * 256 + 256 * 256;
  for (int i = blockIdx.x * blockDim.x + threadIdx.x; i < total;
       i += gridDim.x * blockDim.x) {
    if (i < 768 * 256) {
      int oc = i >> 8, k = i & 255;                 // wqT[oc][k] = qkv_w[k][oc]
      wqT[i] = (__bf16)qkv_w[k * 768 + oc];
    } else {
      int j = i - 768 * 256;
      int oc = j >> 8, k = j & 255;                 // wpT[oc][k] = proj_w[k][oc]
      wpT[j] = (__bf16)proj_w[k * 256 + oc];
    }
  }
}

// -------- main: one block per window, one wave per head --------
__global__ __launch_bounds__(256) void
win_attn_kernel(const float* __restrict__ x, const float* __restrict__ mask,
                const float* __restrict__ qkv_b, const float* __restrict__ proj_b,
                const float* __restrict__ bias_table, const int* __restrict__ rel_index,
                const __bf16* __restrict__ wqT, const __bf16* __restrict__ wpT,
                float* __restrict__ out) {
  extern __shared__ __align__(16) char smemraw[];
  __bf16* S = (__bf16*)smemraw;

  const int tid  = threadIdx.x;
  const int wave = tid >> 5;
  const int lane = tid & 31;
  const int hl   = lane >> 4;    // which 16-lane half
  const int ln   = lane & 15;
  const int b    = blockIdx.x;
  const int w    = b & (NW - 1);
  const int h    = wave;         // head handled by this wave

  __bf16* Xs  = S + OFF_X;
  __bf16* qS  = S + OFF_H + h * HEAD_ELEMS;
  __bf16* kS  = qS + HQ;
  __bf16* vtS = kS + HK;
  __bf16* Pw  = S + OFF_P + wave * (16 * 64);
  __bf16* Ob  = S + OFF_O;

  // ---- stage 0: stage X tile (bf16, zero-pad rows 49..63) ----
  {
    const float* xb = x + (size_t)b * NTOK * CDIM;
    for (int i = tid; i < 64 * CDIM; i += 256) {
      int row = i >> 8, col = i & 255;
      float v = (row < NTOK) ? xb[row * CDIM + col] : 0.0f;
      Xs[row * XSTR + col] = (__bf16)v;
    }
  }
  __syncthreads();

  // ---- stage A: QKV GEMM, this wave produces q/k/v for head h ----
#pragma unroll 1
  for (int nt6 = 0; nt6 < 6; ++nt6) {
    const int s = nt6 >> 1;                               // 0=q 1=k 2=v
    const int colbase = s * CDIM + h * HD + (nt6 & 1) * 16;
    v8f acc[4] = {};
#pragma unroll
    for (int ks = 0; ks < 8; ++ks) {
      // B fragment: column of qkv_w == row of wqT, contiguous in k
      v16bf bf = *(const v16bf*)(wqT + (size_t)(colbase + ln) * CDIM + ks * 32 + hl * 16);
#pragma unroll
      for (int mt = 0; mt < 4; ++mt) {
        const __bf16* ap = Xs + (mt * 16 + ln) * XSTR + ks * 32 + hl * 16;
        v16bf af = frag_split(ap, ap + 8);
        acc[mt] = WMMA_BF16(af, bf, acc[mt]);
      }
    }
    const float bias = qkv_b[colbase + ln];
    const int cl = (nt6 & 1) * 16 + ln;
#pragma unroll
    for (int mt = 0; mt < 4; ++mt) {
#pragma unroll
      for (int r = 0; r < 8; ++r) {
        int row = mt * 16 + r + 8 * hl;
        float v = acc[mt][r] + bias;
        if (s == 0)      qS[row * HD + cl]  = (__bf16)(v * SCALE);
        else if (s == 1) kS[row * HD + cl]  = (__bf16)v;
        else             vtS[cl * VTSTR + row] = (__bf16)v;   // v transposed
      }
    }
  }
  // q/k/vT of head h written and read by this wave only: no barrier needed.

  // ---- stage B: attention for head h ----
  const float* maskw = mask + (size_t)w * NTOK * NTOK;
#pragma unroll 1
  for (int mt = 0; mt < 4; ++mt) {
    const __bf16* ap0 = qS + (mt * 16 + ln) * HD + hl * 16;
    v16bf aq = frag_split(ap0, ap0 + 8);
    v8f sacc[4];
#pragma unroll
    for (int nt = 0; nt < 4; ++nt) {
      v16bf bk = *(const v16bf*)(kS + (nt * 16 + ln) * HD + hl * 16);
      v8f z = {};
      sacc[nt] = WMMA_BF16(aq, bk, z);
    }
    // bias + shift mask; kill padded columns
#pragma unroll
    for (int nt = 0; nt < 4; ++nt) {
      int col = nt * 16 + ln;
#pragma unroll
      for (int r = 0; r < 8; ++r) {
        int row = mt * 16 + r + 8 * hl;
        if (col < NTOK) {
          if (row < NTOK) {
            int ri = rel_index[row * NTOK + col];
            sacc[nt][r] += bias_table[ri * HEADS + h] + maskw[row * NTOK + col];
          }
        } else {
          sacc[nt][r] = -1e30f;
        }
      }
    }
    // softmax per row (rows live in one 16-lane half; xor masks <16 stay inside it)
#pragma unroll
    for (int r = 0; r < 8; ++r) {
      float m = fmaxf(fmaxf(sacc[0][r], sacc[1][r]), fmaxf(sacc[2][r], sacc[3][r]));
#pragma unroll
      for (int off = 8; off >= 1; off >>= 1) m = fmaxf(m, __shfl_xor(m, off, 32));
      float e0 = __expf(sacc[0][r] - m), e1 = __expf(sacc[1][r] - m);
      float e2 = __expf(sacc[2][r] - m), e3 = __expf(sacc[3][r] - m);
      float sum = (e0 + e1) + (e2 + e3);
#pragma unroll
      for (int off = 8; off >= 1; off >>= 1) sum += __shfl_xor(sum, off, 32);
      float inv = 1.0f / sum;
      sacc[0][r] = e0 * inv; sacc[1][r] = e1 * inv;
      sacc[2][r] = e2 * inv; sacc[3][r] = e3 * inv;
    }
    // P tile (16x64) -> per-wave LDS buffer
#pragma unroll
    for (int nt = 0; nt < 4; ++nt)
#pragma unroll
      for (int r = 0; r < 8; ++r)
        Pw[(r + 8 * hl) * 64 + nt * 16 + ln] = (__bf16)sacc[nt][r];

    // O tile = P(16x64) @ V(64x32)
#pragma unroll
    for (int no = 0; no < 2; ++no) {
      v8f oa = {};
#pragma unroll
      for (int ks = 0; ks < 2; ++ks) {
        const __bf16* pp = Pw + ln * 64 + ks * 32 + hl * 16;
        v16bf af = frag_split(pp, pp + 8);
        const __bf16* vp = vtS + (no * 16 + ln) * VTSTR + ks * 32 + hl * 16;
        v16bf bf = frag_split(vp, vp + 8);
        oa = WMMA_BF16(af, bf, oa);
      }
#pragma unroll
      for (int r = 0; r < 8; ++r) {
        int row = mt * 16 + r + 8 * hl;
        Ob[row * OBSTR + h * HD + no * 16 + ln] = (__bf16)oa[r];
      }
    }
  }
  __syncthreads();   // proj reads all heads' O columns

  // ---- stage C: projection, this wave handles output cols [h*32, h*32+32) ----
  float* ob = out + (size_t)b * NTOK * CDIM;
#pragma unroll 1
  for (int nt = 0; nt < 2; ++nt) {
    const int colbase = h * HD + nt * 16;
    v8f acc[4] = {};
#pragma unroll
    for (int ks = 0; ks < 8; ++ks) {
      v16bf bf = *(const v16bf*)(wpT + (size_t)(colbase + ln) * CDIM + ks * 32 + hl * 16);
#pragma unroll
      for (int mt = 0; mt < 4; ++mt) {
        const __bf16* ap = Ob + (mt * 16 + ln) * OBSTR + ks * 32 + hl * 16;
        v16bf af = frag_split(ap, ap + 8);
        acc[mt] = WMMA_BF16(af, bf, acc[mt]);
      }
    }
    const float pb = proj_b[colbase + ln];
#pragma unroll
    for (int mt = 0; mt < 4; ++mt)
#pragma unroll
      for (int r = 0; r < 8; ++r) {
        int row = mt * 16 + r + 8 * hl;
        if (row < NTOK) ob[row * CDIM + colbase + ln] = acc[mt][r] + pb;
      }
  }
}

extern "C" void kernel_launch(void* const* d_in, const int* in_sizes, int n_in,
                              void* d_out, int out_size, void* d_ws, size_t ws_size,
                              hipStream_t stream) {
  const float* x          = (const float*)d_in[0];
  const float* mask       = (const float*)d_in[1];
  const float* qkv_w      = (const float*)d_in[2];
  const float* qkv_b      = (const float*)d_in[3];
  const float* proj_w     = (const float*)d_in[4];
  const float* proj_b     = (const float*)d_in[5];
  const float* bias_table = (const float*)d_in[6];
  const int*   rel_index  = (const int*)d_in[7];
  float* out = (float*)d_out;

  __bf16* wqT = (__bf16*)d_ws;            // 768*256 bf16
  __bf16* wpT = wqT + 768 * 256;          // 256*256 bf16  (total 512 KB of d_ws)

  prep_kernel<<<256, 256, 0, stream>>>(qkv_w, proj_w, wqT, wpT);
  win_attn_kernel<<<BWIN, 256, LDS_BYTES, stream>>>(
      x, mask, qkv_b, proj_b, bias_table, rel_index, wqT, wpT, out);
}
